// GNNTEP_40312563040535
// MI455X (gfx1250) — matrix operand order repaired and checked by reading, not compiled, not run on hardware
//
#include <hip/hip_runtime.h>

// ---------------------------------------------------------------------------
// GNN-TEP for MI455X (gfx1250, wave32, WMMA, async-to-LDS staging).
//   Layouts: activations kept TRANSPOSED [b][w(64)][node(512)] so the
//   aggregation B-operand and all D-tile stores are contiguous b128 ops.
//   AX = (adj ⊙ mask) @ X          once, row-major out   (k_aggr_proj, W=null)
//   per iteration i:
//     Ht  = relu(AX @ W1 + b1)^T   (k_proj, transposed store)
//     stats(Ht) -> alpha,c ; BN folded into adjacency:  adjs = adj·α[m], trow
//     skip = min_n(α·H+c)
//     H2t = relu((adjs @ H + trow) @ W2 + b2)^T  -- single fused kernel
//     stats(H2t) -> alpha,c ; concat[:,i*64+f] = min_n(α·H2+c) + skip
//   out = concat @ Wfc + bfc
// ---------------------------------------------------------------------------

#define NNODES 512
#define NH     64
#define BATCH  1024
#define OUTC   32
#define NG     3

// Flip to 0 if the assembler rejects the async mnemonics.
#define USE_ASYNC_LDS 1

typedef __bf16 bf16_t;
typedef __attribute__((ext_vector_type(16))) __bf16 v16bf;
typedef __attribute__((ext_vector_type(8)))  __bf16 v8bf;
typedef __attribute__((ext_vector_type(8)))  float  v8f;

static __device__ __forceinline__ v8f wmma_bf16(v16bf a, v16bf b, v8f c) {
  return __builtin_amdgcn_wmma_f32_16x16x32_bf16(false, a, false, b, (short)0, c, false, false);
}

static __device__ __forceinline__ void async_cp16(bf16_t* ldsDst, const bf16_t* gSrc) {
#if USE_ASYNC_LDS
  unsigned l = (unsigned)(size_t)ldsDst;   // generic LDS ptr: low 32 bits = LDS offset
  asm volatile("global_load_async_to_lds_b128 %0, %1, off"
               :: "v"(l), "v"(gSrc) : "memory");
#else
  *(uint4*)ldsDst = *(const uint4*)gSrc;
#endif
}
static __device__ __forceinline__ void async_drain() {
#if USE_ASYNC_LDS
  asm volatile("s_wait_asynccnt 0x0" ::: "memory");
#endif
}

static __device__ __forceinline__ v16bf make_afrag(const bf16_t* row, int ks, int hi) {
  // A-frag row `row`, K-chunk ks: lanes hi=0 -> K{ks..+15}, hi=1 -> K{ks+16..+31}
  v8bf a0 = *(const v8bf*)&row[ks + 16 * hi];
  v8bf a1 = *(const v8bf*)&row[ks + 16 * hi + 8];
  return __builtin_shufflevector(a0, a1, 0,1,2,3,4,5,6,7,8,9,10,11,12,13,14,15);
}

// ---------------------------------------------------------------------------
// prep kernels
// ---------------------------------------------------------------------------
__global__ void k_adjmask(const float* __restrict__ adj, bf16_t* __restrict__ adjb) {
  int i = blockIdx.x * blockDim.x + threadIdx.x;
  if (i < NNODES * NNODES) {
    int n = i >> 9, m = i & (NNODES - 1);
    adjb[i] = (bf16_t)((m == n) ? 0.f : adj[i]);
  }
}

__global__ void k_cvt(const float* __restrict__ x, bf16_t* __restrict__ y, long nElem) {
  long i = (long)blockIdx.x * blockDim.x + threadIdx.x;
  if (i < nElem) y[i] = (bf16_t)x[i];
}

// X [b][m(512)][w(64)] f32  ->  Xt [b][w(64)][m(512)] bf16, LDS-tiled transpose
__global__ __launch_bounds__(256) void k_transX(const float* __restrict__ X,
                                                bf16_t* __restrict__ Xt) {
  __shared__ bf16_t Ld[64 * 64];
  const int mc = blockIdx.x;                 // 0..7 (node chunk of 64)
  const int b  = blockIdx.y;
  const int t  = threadIdx.x;
  const long ib = (long)b * (NNODES * NH);
  for (int i = 0; i < 4; ++i) {
    int idx = t * 4 + i * 1024;
    int r = idx >> 6, c = idx & 63;          // r: node-local, c: w
    float4 d = *(const float4*)&X[ib + (long)(mc * 64 + r) * NH + c];
    Ld[r * 64 + c + 0] = (bf16_t)d.x;  Ld[r * 64 + c + 1] = (bf16_t)d.y;
    Ld[r * 64 + c + 2] = (bf16_t)d.z;  Ld[r * 64 + c + 3] = (bf16_t)d.w;
  }
  __syncthreads();
  int w = t >> 2, ms = (t & 3) * 16;
  v8bf o0, o1;
#pragma unroll
  for (int j = 0; j < 8; ++j) { o0[j] = Ld[(ms + j) * 64 + w]; o1[j] = Ld[(ms + 8 + j) * 64 + w]; }
  *(v8bf*)&Xt[ib + (long)w * NNODES + mc * 64 + ms]     = o0;
  *(v8bf*)&Xt[ib + (long)w * NNODES + mc * 64 + ms + 8] = o1;
}

// ---------------------------------------------------------------------------
// k_aggr_proj: per batch b, 64-node tile:
//   C[n][w] = sum_m Aadj[n][m] * Bt[b][w][m]   (+ trow[n])
//   if W:  out_t[b][o][n] = relu(C @ W + bias) (transposed store)
//   else:  out_rm[b][n][w] = C                 (row-major store)
// grid (8, BATCH), block 256 (8 waves); wave -> (nsub = wv>>1, wsub = wv&1).
// ---------------------------------------------------------------------------
__global__ __launch_bounds__(256) void k_aggr_proj(const bf16_t* __restrict__ Aadj,  // [512,512]
                                                   const bf16_t* __restrict__ Bt,    // [B,64,512]
                                                   const float*  __restrict__ trow,  // [512] | null
                                                   const bf16_t* __restrict__ W,     // [64,64] | null
                                                   const float*  __restrict__ bias,  // [64] | null
                                                   bf16_t* __restrict__ outRM,       // [B,512,64] | null
                                                   bf16_t* __restrict__ outT)        // [B,64,512] | null
{
  __shared__ __align__(16) bf16_t As[64 * 64];
  __shared__ __align__(16) bf16_t Bs[64 * 64];
  __shared__ __align__(16) bf16_t Ws[64 * 64];
  __shared__ __align__(16) bf16_t Cs[64 * 64];
  const int t    = threadIdx.x;
  const int lane = t & 31;
  const int wv   = t >> 5;
  const int lo   = lane & 15;
  const int hi   = lane >> 4;
  const int nsub = wv >> 1;                 // 0..3
  const int wsub = wv & 1;                  // 0..1
  const int n0   = blockIdx.x * 64;
  const int b    = blockIdx.y;
  const long bofs = (long)b * (NNODES * NH);

  if (W) {  // stage Wt[n][k] = W[k][n] once
    int k = t >> 2, nb = (t & 3) * 16;
    v8bf w0 = *(const v8bf*)&W[k * 64 + nb];
    v8bf w1 = *(const v8bf*)&W[k * 64 + nb + 8];
#pragma unroll
    for (int j = 0; j < 8; ++j) { Ws[(nb + j) * 64 + k] = w0[j]; Ws[(nb + 8 + j) * 64 + k] = w1[j]; }
  }

  v8f acc0 = {0.f,0.f,0.f,0.f,0.f,0.f,0.f,0.f};
  v8f acc1 = {0.f,0.f,0.f,0.f,0.f,0.f,0.f,0.f};

  for (int kt = 0; kt < NNODES / 64; ++kt) {
    const int k0 = kt * 64;
#pragma unroll
    for (int h = 0; h < 2; ++h) {           // 2 x 16B per thread per array
      int idx = (t + h * 256) * 8;
      int r = idx >> 6, c = idx & 63;
      async_cp16(&As[idx], &Aadj[(long)(n0 + r) * NNODES + k0 + c]);
      async_cp16(&Bs[idx], &Bt[bofs + (long)r * NNODES + k0 + c]);   // r == w here
    }
    async_drain();
    __syncthreads();
#pragma unroll
    for (int s = 0; s < 2; ++s) {
      const int ks = s * 32;
      v16bf a  = make_afrag(&As[(nsub * 16 + lo) * 64], ks, hi);
      v16bf b0 = *(const v16bf*)&Bs[(wsub * 32 + lo) * 64 + ks + 16 * hi];
      v16bf b1 = *(const v16bf*)&Bs[(wsub * 32 + 16 + lo) * 64 + ks + 16 * hi];
      acc0 = wmma_bf16(a, b0, acc0);
      acc1 = wmma_bf16(a, b1, acc1);
    }
    __syncthreads();
  }

  // C tile -> Cs (bf16), with trow
  const int wA = wsub * 32 + lo;
  const int wB = wsub * 32 + 16 + lo;
#pragma unroll
  for (int v = 0; v < 8; ++v) {
    int nl = nsub * 16 + v + 8 * hi;
    float tr = trow ? trow[n0 + nl] : 0.f;
    Cs[nl * 64 + wA] = (bf16_t)(acc0[v] + tr);
    Cs[nl * 64 + wB] = (bf16_t)(acc1[v] + tr);
  }
  __syncthreads();

  if (!W) {  // row-major copy out
#pragma unroll
    for (int h = 0; h < 2; ++h) {
      int idx = (t + h * 256) * 8;
      int r = idx >> 6, c = idx & 63;
      *(uint4*)&outRM[bofs + (long)(n0 + r) * NH + c] = *(const uint4*)&Cs[idx];
    }
    return;
  }

  // fused projection: D = relu(C @ W + bias), store transposed [b][o][n]
  v8f d0 = {0.f,0.f,0.f,0.f,0.f,0.f,0.f,0.f};
  v8f d1 = {0.f,0.f,0.f,0.f,0.f,0.f,0.f,0.f};
#pragma unroll
  for (int s = 0; s < 2; ++s) {
    const int ks = s * 32;
    v16bf a  = make_afrag(&Cs[(nsub * 16 + lo) * 64], ks, hi);
    v16bf b0 = *(const v16bf*)&Ws[wA * 64 + ks + 16 * hi];
    v16bf b1 = *(const v16bf*)&Ws[wB * 64 + ks + 16 * hi];
    d0 = wmma_bf16(a, b0, d0);
    d1 = wmma_bf16(a, b1, d1);
  }
  const float bi0 = bias[wA], bi1 = bias[wB];
  v8bf o0, o1;
#pragma unroll
  for (int v = 0; v < 8; ++v) {
    float x0 = d0[v] + bi0; x0 = x0 > 0.f ? x0 : 0.f; o0[v] = (bf16_t)x0;
    float x1 = d1[v] + bi1; x1 = x1 > 0.f ? x1 : 0.f; o1[v] = (bf16_t)x1;
  }
  const long nbase = n0 + nsub * 16 + 8 * hi;
  *(v8bf*)&outT[bofs + (long)wA * NNODES + nbase] = o0;
  *(v8bf*)&outT[bofs + (long)wB * NNODES + nbase] = o1;
}

// ---------------------------------------------------------------------------
// k_proj (layer 1): Hout_t[b][w][n] = relu(Ain[r,:64] @ W + bias), r=(b,n)
// grid (B*512/16), block 128.
// ---------------------------------------------------------------------------
__global__ __launch_bounds__(128) void k_proj(const bf16_t* __restrict__ Ain,  // [B*512,64] row-major
                                              const bf16_t* __restrict__ Wb,   // [64,64] k-major
                                              const float*  __restrict__ bias,
                                              bf16_t*       __restrict__ HoutT) // [B,64,512]
{
  __shared__ __align__(16) bf16_t Wt[64 * 64];
  const int t    = threadIdx.x;
  const int lane = t & 31;
  const int wv   = t >> 5;
  const int lo   = lane & 15;
  const int hi   = lane >> 4;
  const long r0  = (long)blockIdx.x * 16;
  {
    int k = t >> 1, nb = (t & 1) * 32;
    for (int i = 0; i < 4; ++i) {
      int nn = nb + i * 8;
      v8bf d = *(const v8bf*)&Wb[k * 64 + nn];
#pragma unroll
      for (int j = 0; j < 8; ++j) Wt[(nn + j) * 64 + k] = d[j];
    }
  }
  __syncthreads();

  v8f acc = {0.f,0.f,0.f,0.f,0.f,0.f,0.f,0.f};
#pragma unroll
  for (int kc = 0; kc < 64; kc += 32) {
    v16bf a = make_afrag(&Ain[(r0 + lo) * 64], kc, hi);
    v16bf bf = *(const v16bf*)&Wt[(wv * 16 + lo) * 64 + kc + 16 * hi];
    acc = wmma_bf16(a, bf, acc);
  }
  const int w = wv * 16 + lo;
  const float bw = bias[w];
  v8bf o;
#pragma unroll
  for (int v = 0; v < 8; ++v) {
    float x = acc[v] + bw; x = x > 0.f ? x : 0.f; o[v] = (bf16_t)x;
  }
  const int bb = (int)(r0 >> 9);
  const int nbase = (int)(r0 & 511) + 8 * hi;
  *(v8bf*)&HoutT[((long)bb * NH + w) * NNODES + nbase] = o;
}

// ---------------------------------------------------------------------------
// stats over transposed layout: S[n]+=Σ h, Q[n]+=Σ h²  (S,Q pre-zeroed)
// grid (8 nchunk, 64 bslice), block 256.
// ---------------------------------------------------------------------------
__global__ void k_zero(float* __restrict__ p, int n) {
  int i = blockIdx.x * blockDim.x + threadIdx.x;
  if (i < n) p[i] = 0.f;
}

__global__ __launch_bounds__(256) void k_stats_t(const bf16_t* __restrict__ Ht,
                                                 float* __restrict__ S, float* __restrict__ Q) {
  const int nc = blockIdx.x, sl = blockIdx.y;
  const int t = threadIdx.x;
  const int nl = t & 63, grp = t >> 6;
  const int n = nc * 64 + nl;
  float s = 0.f, q = 0.f;
  for (int it = grp; it < 16 * 64; it += 4) {
    int b = sl * 16 + (it >> 6);
    int w = it & 63;
    float v = (float)Ht[((long)b * NH + w) * NNODES + n];
    s += v; q += v * v;
  }
  __shared__ float shs[256], shq[256];
  shs[t] = s; shq[t] = q;
  __syncthreads();
  if (grp == 0) {
    s = shs[nl] + shs[nl + 64] + shs[nl + 128] + shs[nl + 192];
    q = shq[nl] + shq[nl + 64] + shq[nl + 128] + shq[nl + 192];
    atomicAdd(&S[n], s);
    atomicAdd(&Q[n], q);
  }
}

__global__ void k_bnfin(const float* __restrict__ S, const float* __restrict__ Q,
                        const float* __restrict__ g, const float* __restrict__ be,
                        float* __restrict__ alpha, float* __restrict__ c) {
  int n = blockIdx.x * blockDim.x + threadIdx.x;
  if (n < NNODES) {
    const float inv = 1.0f / (float)(BATCH * NH);
    float mean = S[n] * inv;
    float var  = Q[n] * inv - mean * mean;
    float a = g[n] * rsqrtf(var + 1e-5f);
    alpha[n] = a;
    c[n] = be[n] - mean * a;
  }
}

__global__ __launch_bounds__(256) void k_adjscale(const float* __restrict__ adj,
                                                  const float* __restrict__ alpha,
                                                  const float* __restrict__ c,
                                                  bf16_t* __restrict__ adjs,
                                                  float*  __restrict__ trow) {
  const int n = blockIdx.x;
  const int t = threadIdx.x;
  float tsum = 0.f;
  for (int m = t; m < NNODES; m += 256) {
    float a = (m == n) ? 0.f : adj[n * NNODES + m];
    adjs[n * NNODES + m] = (bf16_t)(a * alpha[m]);
    tsum += a * c[m];
  }
  __shared__ float sh[256];
  sh[t] = tsum;
  __syncthreads();
  for (int off = 128; off > 0; off >>= 1) {
    if (t < off) sh[t] += sh[t + off];
    __syncthreads();
  }
  if (t == 0) trow[n] = sh[0];
}

// out[b, coff+f] = min_n(alpha[n]*Ht[b][f][n]+c[n]) (+ addin[b,f])
__global__ __launch_bounds__(64) void k_minred_t(const bf16_t* __restrict__ Ht,
                                                 const float* __restrict__ alpha,
                                                 const float* __restrict__ c,
                                                 const float* __restrict__ addin,
                                                 float* __restrict__ outp, int ldo, int coff) {
  const int b = blockIdx.x;
  const int f = threadIdx.x;
  const bf16_t* row = &Ht[((long)b * NH + f) * NNODES];
  float m = 3.4e38f;
  for (int n = 0; n < NNODES; n += 8) {
    v8bf v = *(const v8bf*)&row[n];
#pragma unroll
    for (int j = 0; j < 8; ++j) {
      float x = alpha[n + j] * (float)v[j] + c[n + j];
      m = x < m ? x : m;
    }
  }
  if (addin) m += addin[b * NH + f];
  outp[(long)b * ldo + coff + f] = m;
}

__global__ __launch_bounds__(32) void k_fc(const float* __restrict__ cc,
                                           const float* __restrict__ Wfc,
                                           const float* __restrict__ bfc,
                                           float* __restrict__ out) {
  const int b = blockIdx.x;
  const int o = threadIdx.x;
  float s = bfc[o];
  for (int j = 0; j < NG * NH; ++j) s += cc[b * (NG * NH) + j] * Wfc[j * OUTC + o];
  out[b * OUTC + o] = s;
}

// ---------------------------------------------------------------------------
extern "C" void kernel_launch(void* const* d_in, const int* in_sizes, int n_in,
                              void* d_out, int out_size, void* d_ws, size_t ws_size,
                              hipStream_t stream)
{
  const float* X   = (const float*)d_in[0];
  const float* adj = (const float*)d_in[1];
  const float* W1  = (const float*)d_in[2];
  const float* b1  = (const float*)d_in[3];
  const float* g1  = (const float*)d_in[4];
  const float* be1 = (const float*)d_in[5];
  const float* W2  = (const float*)d_in[6];
  const float* b2  = (const float*)d_in[7];
  const float* g2  = (const float*)d_in[8];
  const float* be2 = (const float*)d_in[9];
  const float* Wfc = (const float*)d_in[10];
  const float* bfc = (const float*)d_in[11];
  float* out = (float*)d_out;

  char* ws = (char*)d_ws;
  size_t off = 0;
  auto carve = [&](size_t bytes) -> char* {
    off = (off + 255) & ~(size_t)255;
    char* p = ws + off;
    off += bytes;
    return p;
  };
  const long NBW = (long)BATCH * NNODES * NH;
  bf16_t* Xt    = (bf16_t*)carve(NBW * 2);                     // [B,64,512]
  bf16_t* AX    = (bf16_t*)carve(NBW * 2);                     // [B,512,64] row-major
  bf16_t* Ht    = (bf16_t*)carve(NBW * 2);                     // [B,64,512]
  bf16_t* H2t   = (bf16_t*)carve(NBW * 2);                     // [B,64,512]
  bf16_t* adjb  = (bf16_t*)carve((size_t)NNODES * NNODES * 2);
  bf16_t* adjs  = (bf16_t*)carve((size_t)NNODES * NNODES * 2);
  bf16_t* W1b   = (bf16_t*)carve((size_t)NG * 64 * 64 * 2);
  bf16_t* W2b   = (bf16_t*)carve((size_t)NG * 64 * 64 * 2);
  float*  S     = (float*)carve(NNODES * 4);
  float*  Q     = (float*)carve(NNODES * 4);
  float*  alpha = (float*)carve(NNODES * 4);
  float*  cvec  = (float*)carve(NNODES * 4);
  float*  trow  = (float*)carve(NNODES * 4);
  float*  skip  = (float*)carve((size_t)BATCH * NH * 4);
  float*  ccat  = (float*)carve((size_t)BATCH * NG * NH * 4);

  k_adjmask<<<(NNODES * NNODES + 255) / 256, 256, 0, stream>>>(adj, adjb);
  k_transX<<<dim3(8, BATCH), 256, 0, stream>>>(X, Xt);
  k_cvt<<<(NG * 64 * 64 + 255) / 256, 256, 0, stream>>>(W1, W1b, NG * 64 * 64);
  k_cvt<<<(NG * 64 * 64 + 255) / 256, 256, 0, stream>>>(W2, W2b, NG * 64 * 64);

  // AX = masked-adj @ X (row-major out; reused by all 3 iterations)
  k_aggr_proj<<<dim3(8, BATCH), 256, 0, stream>>>(adjb, Xt, (const float*)nullptr,
                                                  (const bf16_t*)nullptr, (const float*)nullptr,
                                                  AX, (bf16_t*)nullptr);

  const int RBLK = (BATCH * NNODES) / 16;
  for (int i = 0; i < NG; ++i) {
    k_proj<<<RBLK, 128, 0, stream>>>(AX, W1b + i * 4096, b1 + i * 64, Ht);
    k_zero<<<2, 256, 0, stream>>>(S, NNODES);
    k_zero<<<2, 256, 0, stream>>>(Q, NNODES);
    k_stats_t<<<dim3(8, 64), 256, 0, stream>>>(Ht, S, Q);
    k_bnfin<<<2, 256, 0, stream>>>(S, Q, g1 + i * NNODES, be1 + i * NNODES, alpha, cvec);
    k_adjscale<<<NNODES, 256, 0, stream>>>(adj, alpha, cvec, adjs, trow);
    k_minred_t<<<BATCH, 64, 0, stream>>>(Ht, alpha, cvec, (const float*)nullptr, skip, NH, 0);
    // fused: H2t = relu((adjs @ H + trow) @ W2 + b2), transposed out
    k_aggr_proj<<<dim3(8, BATCH), 256, 0, stream>>>(adjs, Ht, trow,
                                                    W2b + i * 4096, b2 + i * 64,
                                                    (bf16_t*)nullptr, H2t);
    k_zero<<<2, 256, 0, stream>>>(S, NNODES);
    k_zero<<<2, 256, 0, stream>>>(Q, NNODES);
    k_stats_t<<<dim3(8, 64), 256, 0, stream>>>(H2t, S, Q);
    k_bnfin<<<2, 256, 0, stream>>>(S, Q, g2 + i * NNODES, be2 + i * NNODES, alpha, cvec);
    k_minred_t<<<BATCH, 64, 0, stream>>>(H2t, alpha, cvec, skip, ccat, NG * NH, i * 64);
  }

  k_fc<<<BATCH, 32, 0, stream>>>(ccat, Wfc, bfc, out);
}
